// MutiHeadAttn_87290915324372
// MI455X (gfx1250) — compile-verified, matrix operand order checked by reading, not run on hardware
//
#include <hip/hip_runtime.h>
#include <hip/hip_bf16.h>

// ---------------------------------------------------------------------------
// Types for CDNA5 WMMA (wave32): v_wmma_f32_16x16x32_bf16
// ---------------------------------------------------------------------------
typedef __bf16 bhalf;
typedef bhalf bhalf16 __attribute__((ext_vector_type(16)));
typedef float f32x8  __attribute__((ext_vector_type(8)));

union Frag {           // 16 bf16 = 8 dwords per lane (A or B operand)
    bhalf16  v;
    unsigned u[8];
};

__device__ __forceinline__ unsigned short f2bf(float f) {
    unsigned u = __builtin_bit_cast(unsigned, f);
    u = u + 0x7FFFu + ((u >> 16) & 1u);      // round-to-nearest-even
    return (unsigned short)(u >> 16);
}
__device__ __forceinline__ float bf2f(unsigned short h) {
    unsigned u = ((unsigned)h) << 16;
    return __builtin_bit_cast(float, u);
}

// ---------------------------------------------------------------------------
// Transpose + convert: W (K x N fp32, row-major, batched) ->
//                      Wt (Npad x K bf16, row-major), zero-filled n >= N.
// ---------------------------------------------------------------------------
__global__ __launch_bounds__(256)
void cvt_transpose_kernel(const float* __restrict__ src,
                          unsigned short* __restrict__ dst,
                          int K, int N, int Npad) {
    __shared__ float tile[32][33];
    const float* s = src + (size_t)blockIdx.z * K * N;
    unsigned short* d = dst + (size_t)blockIdx.z * Npad * K;
    int k0 = blockIdx.x * 32, n0 = blockIdx.y * 32;
    int tx = threadIdx.x & 31, ty = threadIdx.x >> 5;
    for (int i = ty; i < 32; i += 8) {
        int n = n0 + tx;
        tile[i][tx] = (n < N) ? s[(size_t)(k0 + i) * N + n] : 0.0f;
    }
    __syncthreads();
    for (int i = ty; i < 32; i += 8)                      // row n0+i of Wt
        d[(size_t)(n0 + i) * K + k0 + tx] = f2bf(tile[tx][i]);
}

// ---------------------------------------------------------------------------
// Gather topic rows -> bf16 X (25600 x 512). One dword (2 elems) per thread.
// ---------------------------------------------------------------------------
__global__ __launch_bounds__(256)
void gather_kernel(const float* __restrict__ topic,
                   const int* __restrict__ seq,
                   unsigned short* __restrict__ Xg) {
    int row = blockIdx.x;                    // 0..25599  (= b*50 + l)
    int idx = seq[row];
    const float2* src = (const float2*)(topic + (size_t)idx * 512);
    unsigned* dst = (unsigned*)(Xg + (size_t)row * 512);
    float2 f = src[threadIdx.x];
    dst[threadIdx.x] = (unsigned)f2bf(f.x) | ((unsigned)f2bf(f.y) << 16);
}

// ---------------------------------------------------------------------------
// bf16 WMMA GEMM: C = act(A(MxK) @ W(KxN) + bias), weights pre-transposed:
//   Wt is (Npad x K) bf16 row-major, Npad = gridDim.x*128.
// M multiple of 64, K multiple of 32 -> staging is guard-free b128.
// Double-buffered LDS pipeline: one barrier per K-step, next slab's global
// loads in flight during the WMMAs.
// Batched over gridDim.z via strides (bsW / bsB / bsC, in elements).
// ---------------------------------------------------------------------------
__global__ __launch_bounds__(256)
void gemm_bf16_kernel(const unsigned short* __restrict__ A, int lda,
                      const unsigned short* __restrict__ Wt,
                      const float* __restrict__ bias,
                      float* __restrict__ C32,
                      unsigned short* __restrict__ C16, int ldc,
                      int Nstore, int K, int relu,
                      size_t bsW, size_t bsB, size_t bsC) {
    __shared__ unsigned aS[2][64 * 16];   // [buf][row][kword]  32 K = 16 dwords
    __shared__ unsigned wS[2][128 * 16];  // [buf][col][kword]  (Wt rows)

    const int tid  = threadIdx.x;
    const int wave = tid >> 5, lane = tid & 31;
    const int wm = wave >> 2, wn = wave & 3;
    const int lh = lane >> 4, ll = lane & 15;
    const int m0 = blockIdx.y * 64;
    const int n0 = blockIdx.x * 128;
    const int z  = blockIdx.z;

    const unsigned short* Wz = Wt + (size_t)z * bsW;
    // guard-free staging (uint4 = 8 bf16); one K-step = 4 uint4 per row
    const int srow = tid >> 2, squad = tid & 3;          // 64 rows x 4 quads
    const uint4* Ap  = (const uint4*)(A  + (size_t)(m0 + srow) * lda) + squad;
    const uint4* Wp0 = (const uint4*)(Wz + (size_t)(n0 + srow)      * K) + squad;
    const uint4* Wp1 = (const uint4*)(Wz + (size_t)(n0 + srow + 64) * K) + squad;
    const int sidx = srow * 16 + squad * 4;

    const int ksteps = K >> 5;
    uint4 ra = Ap[0], rw0 = Wp0[0], rw1 = Wp1[0];
    *(uint4*)&aS[0][sidx] = ra;
    *(uint4*)&wS[0][sidx] = rw0;
    *(uint4*)&wS[0][sidx + 64 * 16] = rw1;

    f32x8 acc[2][2] = {};
    for (int ks = 0; ks < ksteps; ++ks) {
        const int cur = ks & 1;
        __syncthreads();
        if (ks + 1 < ksteps) {                 // next slab -> regs (in flight)
            ra  = Ap [(ks + 1) * 4];
            rw0 = Wp0[(ks + 1) * 4];
            rw1 = Wp1[(ks + 1) * 4];
            __builtin_prefetch(Ap  + (ks + 2) * 4, 0, 0);  // speculative
            __builtin_prefetch(Wp0 + (ks + 2) * 4, 0, 0);
            __builtin_prefetch(Wp1 + (ks + 2) * 4, 0, 0);
        }

        const unsigned* aB = aS[cur];
        const unsigned* wB = wS[cur];
        Frag af[2], bf[2];
        #pragma unroll
        for (int mt = 0; mt < 2; ++mt) {
            const unsigned* ap = &aB[(wm * 32 + mt * 16 + ll) * 16 + lh * 4];
            #pragma unroll
            for (int v = 0; v < 4; ++v) { af[mt].u[v] = ap[v]; af[mt].u[v + 4] = ap[v + 8]; }
        }
        #pragma unroll
        for (int nt = 0; nt < 2; ++nt) {
            const unsigned* bp = &wB[(wn * 32 + nt * 16 + ll) * 16 + lh * 4];
            #pragma unroll
            for (int v = 0; v < 4; ++v) { bf[nt].u[v] = bp[v]; bf[nt].u[v + 4] = bp[v + 8]; }
        }
        #pragma unroll
        for (int mt = 0; mt < 2; ++mt)
            #pragma unroll
            for (int nt = 0; nt < 2; ++nt)
                acc[mt][nt] = __builtin_amdgcn_wmma_f32_16x16x32_bf16(
                    false, af[mt].v, false, bf[nt].v, (short)0, acc[mt][nt],
                    false, false);

        if (ks + 1 < ksteps) {                 // fill the alternate buffer
            *(uint4*)&aS[cur ^ 1][sidx] = ra;
            *(uint4*)&wS[cur ^ 1][sidx] = rw0;
            *(uint4*)&wS[cur ^ 1][sidx + 64 * 16] = rw1;
        }
    }

    // epilogue: C layout = VGPR r -> row (r + 8*lh), col = ll  (ISA 7.12.2)
    const float* biasz = bias ? bias + (size_t)z * bsB : nullptr;
    float*          c32z = C32 ? C32 + (size_t)z * bsC : nullptr;
    unsigned short* c16z = C16 ? C16 + (size_t)z * bsC : nullptr;
    #pragma unroll
    for (int mt = 0; mt < 2; ++mt)
        #pragma unroll
        for (int nt = 0; nt < 2; ++nt) {
            int n = n0 + wn * 32 + nt * 16 + ll;
            if (n >= Nstore) continue;
            float bv = biasz ? biasz[n] : 0.0f;
            #pragma unroll
            for (int r = 0; r < 8; ++r) {
                int m = m0 + wm * 32 + mt * 16 + lh * 8 + r;
                float val = acc[mt][nt][r] + bv;
                if (relu) val = fmaxf(val, 0.0f);
                if (c32z) c32z[(size_t)m * ldc + n] = val;
                if (c16z) c16z[(size_t)m * ldc + n] = f2bf(val);
            }
        }
}

// ---------------------------------------------------------------------------
// Extract ht = fea[:, 49, :]  (bf16)
// ---------------------------------------------------------------------------
__global__ void ht_kernel(const unsigned short* __restrict__ fea16,
                          unsigned short* __restrict__ ht16) {
    int b = blockIdx.x, h = threadIdx.x;
    ht16[(size_t)b * 256 + h] = fea16[((size_t)b * 50 + 49) * 256 + h];
}

// ---------------------------------------------------------------------------
// Fused attention kernel per (b, n):
//   q2 = fea_b @ w2_n + w2b_n           (WMMA, never leaves LDS/regs)
//   alpha[l] = sum_d sigmoid(q1+q2)*v   (in-register, shfl half-wave reduce)
//   a[b,n,:] = sum_l alpha*mask*fea     -> c_final (with ht interleaved)
// w2t is pre-transposed (8, 256(d), 256(h)) bf16; double-buffered K slabs.
// ---------------------------------------------------------------------------
__global__ __launch_bounds__(256)
void attn_kernel(const unsigned short* __restrict__ fea16,  // (B*50, 256)
                 const unsigned short* __restrict__ w2t,    // (8,256,256) [n][d][h]
                 const float* __restrict__ q1,              // (B,8,256)
                 const float* __restrict__ w2b,             // (8,256)
                 const float* __restrict__ v_w,             // (8,256)
                 const int* __restrict__ seq,               // (B,50)
                 unsigned short* __restrict__ cfin) {       // (B,4096)
    __shared__ unsigned feaS[64 * 128];    // fea rows (50 real + 14 zero pad)
    __shared__ unsigned w2S[2][256 * 16];  // double-buffered weight K-slab
    __shared__ float    alphaS[64];

    const int n = blockIdx.x, b = blockIdx.y;
    const int tid = threadIdx.x;
    const int wave = tid >> 5, lane = tid & 31;
    const int lh = lane >> 4, ll = lane & 15;

    // load fea block (1600 uint4) + zero pad rows 50..63 (448 uint4)
    const uint4* feaG = (const uint4*)(fea16 + (size_t)b * 50 * 256);
    uint4* feaD = (uint4*)feaS;
    for (int i = tid; i < 1600; i += 256) feaD[i] = feaG[i];
    for (int i = tid; i < 448; i += 256) feaD[1600 + i] = make_uint4(0, 0, 0, 0);
    if (tid < 64) alphaS[tid] = 0.0f;

    // w2t row per thread: 32 uint4 total, 4 per K-step
    const uint4* Wp = (const uint4*)(w2t + ((size_t)n * 256 + tid) * 256);
    uint4 rw[4];
    #pragma unroll
    for (int q = 0; q < 4; ++q) rw[q] = Wp[q];
    {
        uint4* d0 = (uint4*)&w2S[0][tid * 16];
        #pragma unroll
        for (int q = 0; q < 4; ++q) d0[q] = rw[q];
    }

    f32x8 acc[4][2] = {};      // 4 M-tiles (rows 0..63 cover L=50) x 2 N-tiles

    for (int ks = 0; ks < 8; ++ks) {           // kb = 32*ks
        const int cur = ks & 1;
        __syncthreads();
        if (ks + 1 < 8) {
            #pragma unroll
            for (int q = 0; q < 4; ++q) rw[q] = Wp[(ks + 1) * 4 + q];
        }

        Frag af[4], bfr[2];
        #pragma unroll
        for (int mt = 0; mt < 4; ++mt) {
            const unsigned* ap = &feaS[(mt * 16 + ll) * 128 + ks * 16 + lh * 4];
            #pragma unroll
            for (int v = 0; v < 4; ++v) { af[mt].u[v] = ap[v]; af[mt].u[v + 4] = ap[v + 8]; }
        }
        #pragma unroll
        for (int j = 0; j < 2; ++j) {
            const unsigned* bp = &w2S[cur][((wave * 2 + j) * 16 + ll) * 16 + lh * 4];
            #pragma unroll
            for (int v = 0; v < 4; ++v) { bfr[j].u[v] = bp[v]; bfr[j].u[v + 4] = bp[v + 8]; }
        }
        #pragma unroll
        for (int mt = 0; mt < 4; ++mt)
            #pragma unroll
            for (int j = 0; j < 2; ++j)
                acc[mt][j] = __builtin_amdgcn_wmma_f32_16x16x32_bf16(
                    false, af[mt].v, false, bfr[j].v, (short)0, acc[mt][j],
                    false, false);

        if (ks + 1 < 8) {
            uint4* d1 = (uint4*)&w2S[cur ^ 1][tid * 16];
            #pragma unroll
            for (int q = 0; q < 4; ++q) d1[q] = rw[q];
        }
    }

    // alpha = sum_d sigmoid(q1 + q2 + w2_b) * v
    #pragma unroll
    for (int j = 0; j < 2; ++j) {
        int col = (wave * 2 + j) * 16 + ll;
        float q1v = q1[((size_t)b * 8 + n) * 256 + col] + w2b[n * 256 + col];
        float vv  = v_w[n * 256 + col];
        #pragma unroll
        for (int mt = 0; mt < 4; ++mt) {
            #pragma unroll
            for (int r = 0; r < 8; ++r) {
                int m = mt * 16 + lh * 8 + r;
                float x = acc[mt][j][r] + q1v;
                float s = 1.0f / (1.0f + __expf(-x));
                float p = (m < 50) ? s * vv : 0.0f;
                #pragma unroll
                for (int off = 1; off < 16; off <<= 1)
                    p += __shfl_xor(p, off, 32);   // halves reduce independently
                if (ll == 0) atomicAdd(&alphaS[m], p);   // ds_add_f32
            }
        }
    }
    __syncthreads();
    if (tid < 50) { if (!(seq[b * 50 + tid] > 0)) alphaS[tid] = 0.0f; }
    __syncthreads();

    // a[b,n,h] and ht interleave into c_final
    {
        int t = tid;                       // h index 0..255
        float av = 0.0f;
        for (int l = 0; l < 50; ++l) {
            unsigned w = feaS[l * 128 + (t >> 1)];
            unsigned short hb = (t & 1) ? (unsigned short)(w >> 16)
                                        : (unsigned short)(w & 0xffffu);
            av += alphaS[l] * bf2f(hb);
        }
        size_t base = (size_t)b * 4096 + (size_t)n * 512;
        cfin[base + t] = f2bf(av);
        unsigned w = feaS[49 * 128 + (t >> 1)];          // ht = fea[:, -1, :]
        cfin[base + 256 + t] = (t & 1) ? (unsigned short)(w >> 16)
                                       : (unsigned short)(w & 0xffffu);
    }
}

// ---------------------------------------------------------------------------
// BatchNorm (over batch of 512) + ReLU.  One thread per column.
// ---------------------------------------------------------------------------
__global__ __launch_bounds__(256)
void bn_relu_kernel(const float* __restrict__ z, const float* __restrict__ g,
                    const float* __restrict__ be, float* __restrict__ h32,
                    unsigned short* __restrict__ h16) {
    int j = threadIdx.x;
    float s = 0.0f, s2 = 0.0f;
    for (int i = 0; i < 512; ++i) {
        float v = z[(size_t)i * 256 + j];
        s += v; s2 += v * v;
    }
    float mean = s * (1.0f / 512.0f);
    float var  = s2 * (1.0f / 512.0f) - mean * mean;
    float scale = __frsqrt_rn(var + 1e-5f) * g[j];
    float shift = be[j];
    for (int i = 0; i < 512; ++i) {
        float v = (z[(size_t)i * 256 + j] - mean) * scale + shift;
        v = fmaxf(v, 0.0f);
        h32[(size_t)i * 256 + j] = v;
        h16[(size_t)i * 256 + j] = f2bf(v);
    }
}

// ---------------------------------------------------------------------------
// hc = bf16(h + c)
// ---------------------------------------------------------------------------
__global__ void addc_kernel(const float* __restrict__ h,
                            const float* __restrict__ c,
                            unsigned short* __restrict__ hc, int nelem) {
    int i = blockIdx.x * 256 + threadIdx.x;
    if (i < nelem) hc[i] = f2bf(h[i] + c[i]);
}

// ---------------------------------------------------------------------------
// Host orchestration
// ---------------------------------------------------------------------------
extern "C" void kernel_launch(void* const* d_in, const int* in_sizes, int n_in,
                              void* d_out, int out_size, void* d_ws, size_t ws_size,
                              hipStream_t stream) {
    (void)in_sizes; (void)n_in; (void)out_size; (void)ws_size;
    const int*   seq    = (const int*)  d_in[0];
    const float* topic  = (const float*)d_in[1];
    const float* dr_w   = (const float*)d_in[2];
    const float* dr_b   = (const float*)d_in[3];
    const float* w1_w   = (const float*)d_in[4];
    const float* w1_b   = (const float*)d_in[5];
    const float* w2_w   = (const float*)d_in[6];
    const float* w2_b   = (const float*)d_in[7];
    const float* v_w    = (const float*)d_in[8];
    const float* lin0_w = (const float*)d_in[9];
    const float* lin0_b = (const float*)d_in[10];
    const float* l_w[3]  = {(const float*)d_in[11], (const float*)d_in[15], (const float*)d_in[19]};
    const float* l_b[3]  = {(const float*)d_in[12], (const float*)d_in[16], (const float*)d_in[20]};
    const float* l_g[3]  = {(const float*)d_in[13], (const float*)d_in[17], (const float*)d_in[21]};
    const float* l_be[3] = {(const float*)d_in[14], (const float*)d_in[18], (const float*)d_in[22]};
    const float* out_w  = (const float*)d_in[23];
    float* out = (float*)d_out;

    const int NWP = 50048;               // 49999 padded to multiple of 128

    // bump allocator over workspace
    size_t off = 0;
    char* ws = (char*)d_ws;
    auto alloc = [&](size_t bytes) -> char* {
        char* p = ws + off;
        off += (bytes + 255) & ~(size_t)255;
        return p;
    };
    unsigned short* drwT   = (unsigned short*)alloc((size_t)256 * 512 * 2);   // (256,512)
    unsigned short* w1T    = (unsigned short*)alloc((size_t)8 * 256 * 256 * 2);
    unsigned short* w2T    = (unsigned short*)alloc((size_t)8 * 256 * 256 * 2);
    unsigned short* lin0T  = (unsigned short*)alloc((size_t)256 * 4096 * 2);
    unsigned short* lT[3];
    for (int i = 0; i < 3; ++i) lT[i] = (unsigned short*)alloc((size_t)256 * 256 * 2);
    unsigned short* outwT  = (unsigned short*)alloc((size_t)NWP * 256 * 2);
    unsigned short* Xg16   = (unsigned short*)alloc((size_t)25600 * 512 * 2);
    unsigned short* fea16  = (unsigned short*)alloc((size_t)25600 * 256 * 2);
    unsigned short* ht16   = (unsigned short*)alloc((size_t)512 * 256 * 2);
    float*          q1     = (float*)alloc((size_t)512 * 8 * 256 * 4);
    unsigned short* cfin16 = (unsigned short*)alloc((size_t)512 * 4096 * 2);
    float*          c32    = (float*)alloc((size_t)512 * 256 * 4);
    unsigned short* c16    = (unsigned short*)alloc((size_t)512 * 256 * 2);
    float*          z32    = (float*)alloc((size_t)512 * 256 * 4);
    float*          h32    = (float*)alloc((size_t)512 * 256 * 4);
    unsigned short* h16    = (unsigned short*)alloc((size_t)512 * 256 * 2);
    unsigned short* hc16   = (unsigned short*)alloc((size_t)512 * 256 * 2);

    // weight transposes (K x N fp32 -> Npad x K bf16)
    cvt_transpose_kernel<<<dim3(512/32, 256/32, 1), dim3(256), 0, stream>>>(dr_w,   drwT,  512,  256,  256);
    cvt_transpose_kernel<<<dim3(256/32, 256/32, 8), dim3(256), 0, stream>>>(w1_w,   w1T,   256,  256,  256);
    cvt_transpose_kernel<<<dim3(256/32, 256/32, 8), dim3(256), 0, stream>>>(w2_w,   w2T,   256,  256,  256);
    cvt_transpose_kernel<<<dim3(4096/32, 256/32, 1), dim3(256), 0, stream>>>(lin0_w, lin0T, 4096, 256,  256);
    for (int i = 0; i < 3; ++i)
        cvt_transpose_kernel<<<dim3(256/32, 256/32, 1), dim3(256), 0, stream>>>(l_w[i], lT[i], 256, 256, 256);
    cvt_transpose_kernel<<<dim3(256/32, NWP/32, 1), dim3(256), 0, stream>>>(out_w, outwT, 256, 49999, NWP);

    // gather topic rows -> bf16
    gather_kernel<<<dim3(25600), dim3(256), 0, stream>>>(topic, seq, Xg16);

    // fea = relu(X @ dr_w + dr_b)                 M=25600 K=512 N=256
    gemm_bf16_kernel<<<dim3(2, 400, 1), dim3(256), 0, stream>>>(
        Xg16, 512, drwT, dr_b, nullptr, fea16, 256, 256, 512, 1, 0, 0, 0);

    ht_kernel<<<dim3(512), dim3(256), 0, stream>>>(fea16, ht16);

    // q1[:,n,:] = ht @ w1_w[n] + w1_b[n]   batched over z: M=512 K=256 N=256
    gemm_bf16_kernel<<<dim3(2, 8, 8), dim3(256), 0, stream>>>(
        ht16, 256, w1T, w1_b, q1, nullptr, 2048, 256, 256, 0,
        (size_t)256 * 256, 256, 256);

    // fused attention -> c_final
    attn_kernel<<<dim3(8, 512), dim3(256), 0, stream>>>(
        fea16, w2T, q1, w2_b, v_w, seq, cfin16);

    // c = relu(c_final @ lin0_w + lin0_b)         M=512 K=4096 N=256
    gemm_bf16_kernel<<<dim3(2, 8, 1), dim3(256), 0, stream>>>(
        cfin16, 4096, lin0T, lin0_b, c32, c16, 256, 256, 4096, 1, 0, 0, 0);

    // 3x (GEMM -> BN -> ReLU)
    const unsigned short* cur = c16;
    for (int i = 0; i < 3; ++i) {
        gemm_bf16_kernel<<<dim3(2, 8, 1), dim3(256), 0, stream>>>(
            cur, 256, lT[i], l_b[i], z32, nullptr, 256, 256, 256, 0, 0, 0, 0);
        bn_relu_kernel<<<dim3(1), dim3(256), 0, stream>>>(z32, l_g[i], l_be[i], h32, h16);
        cur = h16;
    }

    // hc = bf16(h + c);  out = hc @ out_w          M=512 K=256 N=49999
    addc_kernel<<<dim3(512), dim3(256), 0, stream>>>(h32, c32, hc16, 512 * 256);
    gemm_bf16_kernel<<<dim3(NWP / 128, 8, 1), dim3(256), 0, stream>>>(
        hc16, 256, outwT, nullptr, out, nullptr, 49999, 49999, 256, 0, 0, 0, 0);
}
